// HAM_35785667510531
// MI455X (gfx1250) — matrix operand order repaired and checked by reading, not compiled, not run on hardware
//
#include <hip/hip_runtime.h>
#include <cmath>

typedef _Float16 h8_t  __attribute__((ext_vector_type(8)));
typedef _Float16 h16_t __attribute__((ext_vector_type(16)));
typedef float    f8_t  __attribute__((ext_vector_type(8)));

#if __has_builtin(__builtin_amdgcn_tanhf)
#define TANHF(x) __builtin_amdgcn_tanhf(x)
#else
#define TANHF(x) tanhf(x)
#endif

#define DHID 256
#define ROWS_PER_BLOCK 64
#define NTHREADS 256   // 8 waves: wave pair (g = wv>>1) shares 16 rows, p = wv&1 picks 8 N-tiles

// ---------------------------------------------------------------------------
// LDS operand layouts (all indices in _Float16 units):
//   B operand (32x16 tile for v_wmma_*_16x16x32): element i of lane L is
//     B[kk*32 + (i>>3)*16 + (L>>4)*8 + (i&7)][t*16 + (L&15)], i = part*8 + j.
//     BIDX(t,kk,part,L,j) = ((t*8+kk)*2+part)*256 + L*8 + j
//   A operand (16x32 tile): element i of lane L is
//     G[L&15][kk*32 + (i>>3)*16 + (L>>4)*8 + (i&7)]
//     AIDX(g,kk,part,L,j) = g*4096 + kk*512 + part*256 + L*8 + j
// Both give a 16B lane stride per ds_load_b128 -> conflict-free LDS reads.
// ---------------------------------------------------------------------------

// x[tt] += G(16x256) @ B(256x16) for this wave's 8 N-tiles (t = p*8+tt).
__device__ __forceinline__ void gemm_acc(const _Float16* __restrict__ sGg,  // already + g*4096
                                         const _Float16* __restrict__ sB,
                                         f8_t x[8], int p, int lane)
{
#pragma unroll
    for (int kk = 0; kk < 8; ++kk) {
        const _Float16* ap = sGg + kk * 512 + lane * 8;
        h8_t a0 = *(const h8_t*)(ap);
        h8_t a1 = *(const h8_t*)(ap + 256);
        h16_t a = __builtin_shufflevector(a0, a1, 0,1,2,3,4,5,6,7,8,9,10,11,12,13,14,15);
#pragma unroll
        for (int tt = 0; tt < 8; ++tt) {
            const int t = p * 8 + tt;
            const _Float16* bp = sB + (t * 8 + kk) * 512 + lane * 8;
            h8_t b0 = *(const h8_t*)(bp);
            h8_t b1 = *(const h8_t*)(bp + 256);
            h16_t b = __builtin_shufflevector(b0, b1, 0,1,2,3,4,5,6,7,8,9,10,11,12,13,14,15);
            x[tt] = __builtin_amdgcn_wmma_f32_16x16x32_f16(
                false, a, false, b, (short)0, x[tt], false, false);
        }
    }
}

__global__ __launch_bounds__(NTHREADS, 1)
void ham_fused_kernel(const float* __restrict__ xv_in,
                      const float* __restrict__ xh_in,
                      const float* __restrict__ W,
                      const int* __restrict__ iters_p,
                      float* __restrict__ out)
{
    // 320 KB total, all in WMMA-operand order.
    __shared__ __align__(16) _Float16 sBW [DHID * DHID];   // W   as B-operand (for Gv @ W)
    __shared__ __align__(16) _Float16 sBWT[DHID * DHID];   // W^T as B-operand (for Gh @ W^T)
    __shared__ __align__(16) _Float16 sGv[4 * 4096];       // tanh(xv) as A-operand, 4 row-groups
    __shared__ __align__(16) _Float16 sGh[4 * 4096];       // tanh(xh) as A-operand

    const int tid  = threadIdx.x;
    const int lane = tid & 31;
    const int wv   = tid >> 5;
    const int g    = wv >> 1;       // row-group (16 rows) shared by a wave pair
    const int p    = wv & 1;        // which 8 of the 16 N-tiles this wave owns
    const int half = lane >> 4;
    const int nloc = lane & 15;
    const int iters = iters_p[0];

    // ---- One-time: W (f32) -> f16 LDS in both B-operand layouts ----
    for (int i = tid; i < DHID * DHID; i += NTHREADS) {
        const int k = i >> 8;       // W row   (contracted in Gv @ W)
        const int n = i & 255;      // W col
        const _Float16 hw = (_Float16)W[i];
        // B = W:    tile t=n>>4, kk=k>>5, part=(k>>4)&1, L=((k>>3)&1)*16+(n&15), j=k&7
        sBW [(n >> 4) * 4096 + (k >> 5) * 512 + ((k >> 4) & 1) * 256 +
             ((k >> 3) & 1) * 128 + (n & 15) * 8 + (k & 7)] = hw;
        // B = W^T:  B2[k'=n][n'=k]
        sBWT[(k >> 4) * 4096 + (n >> 5) * 512 + ((n >> 4) & 1) * 256 +
             ((n >> 3) & 1) * 128 + (k & 15) * 8 + (n & 7)] = hw;
    }

    // ---- State in registers, WMMA C/D layout: lane covers col c=(p*8+tt)*16+nloc,
    //      rows rowBase..rowBase+7 (rowBase includes half*8). 128 VGPRs/wave. ----
    f8_t xv[8], xh[8];
    const int rowBase = blockIdx.x * ROWS_PER_BLOCK + g * 16 + half * 8;
#pragma unroll
    for (int tt = 0; tt < 8; ++tt) {
#pragma unroll
        for (int i = 0; i < 8; ++i) {
            const int idx = (rowBase + i) * DHID + (p * 8 + tt) * 16 + nloc;
            xv[tt][i] = xv_in[idx];
            xh[tt][i] = xh_in[idx];
        }
    }
    __syncthreads();

    // Lane-constant base for scattering tanh(x) into the A-operand layout:
    // AIDX for (row r = half*8+i, col c = t*16+nloc)
    //   = g*4096 + t*256 + (nloc>>3)*128 + half*64 + i*8 + (nloc&7)
    const int gBase = g * 4096 + (nloc >> 3) * 128 + half * 64 + (nloc & 7);
    _Float16* gvB = sGv + gBase;
    _Float16* ghB = sGh + gBase;

    // ---- Energy-descent iterations, fully resident ----
    for (int it = 0; it < iters; ++it) {
        // Snapshot g = tanh(x) (this wave covers its 8 column-tiles of its 16 rows;
        // the partner wave covers the other 8 -> full K for the A-operand).
#pragma unroll
        for (int tt = 0; tt < 8; ++tt) {
            const int toff = (p * 8 + tt) * 256;
#pragma unroll
            for (int i = 0; i < 8; ++i) {
                gvB[toff + i * 8] = (_Float16)TANHF(xv[tt][i]);
                ghB[toff + i * 8] = (_Float16)TANHF(xh[tt][i]);
            }
        }
        __syncthreads();

        // xh <- 0.5*(xh + Gv @ W);  xv <- 0.5*(xv + Gh @ W^T)
        gemm_acc(sGv + g * 4096, sBW,  xh, p, lane);
        gemm_acc(sGh + g * 4096, sBWT, xv, p, lane);

#pragma unroll
        for (int tt = 0; tt < 8; ++tt) {
            xh[tt] *= 0.5f;
            xv[tt] *= 0.5f;
        }
        __syncthreads();  // protect sGv/sGh before next snapshot
    }

    // ---- Output: [tanh(xv), tanh(xh)] along dim 1 (row width 512) ----
#pragma unroll
    for (int tt = 0; tt < 8; ++tt) {
#pragma unroll
        for (int i = 0; i < 8; ++i) {
            const int r = rowBase + i;
            const int c = (p * 8 + tt) * 16 + nloc;
            out[r * (2 * DHID) + c]        = TANHF(xv[tt][i]);
            out[r * (2 * DHID) + DHID + c] = TANHF(xh[tt][i]);
        }
    }
}

extern "C" void kernel_launch(void* const* d_in, const int* in_sizes, int n_in,
                              void* d_out, int out_size, void* d_ws, size_t ws_size,
                              hipStream_t stream) {
    const float* xv = (const float*)d_in[0];
    const float* xh = (const float*)d_in[1];
    const float* W  = (const float*)d_in[2];
    const int*   it = (const int*)d_in[3];
    float* out = (float*)d_out;

    const int rows = in_sizes[0] / DHID;           // 32768
    dim3 grid((unsigned)(rows / ROWS_PER_BLOCK));  // 512 blocks
    ham_fused_kernel<<<grid, NTHREADS, 0, stream>>>(xv, xh, W, it, out);
}
